// Attention_20658792694682
// MI455X (gfx1250) — matrix operand order repaired and checked by reading, not compiled, not run on hardware
//
#include <hip/hip_runtime.h>

// ---------------------------------------------------------------------------
// CDNA5 (gfx1250) attention, f16 WMMA with f32 accumulation.
//   K0: fp32 -> f16 converts (x, qkv_w, proj_w)
//   K1: QKV GEMM  (8192x768 @ 768x2304) -> Q(scaled), K  [B,H,N,D], V^T [B,H,D,N]
//   K2: flash attention per (b,h), online softmax, WMMA for QK^T and PV
//   K3: output projection GEMM + bias -> fp32 d_out
// Everything lives in L2 (192MB) so register-tiled WMMA w/ direct loads is fine.
// Workspace: ~68 MB of f16.
// ---------------------------------------------------------------------------

typedef __attribute__((ext_vector_type(8)))  _Float16 half8;
typedef __attribute__((ext_vector_type(16))) _Float16 half16;
typedef __attribute__((ext_vector_type(8)))  float    float8;

__device__ __forceinline__ float8 wmma16(half16 a, half16 b, float8 c) {
  // (neg_a, A, neg_b, B, c_mod, C, reuse_a, reuse_b)
  return __builtin_amdgcn_wmma_f32_16x16x32_f16(false, a, false, b, (short)0, c, false, false);
}

// A-fragment 16x32 f16: lane holds row (lane&15); halves: k = 8*kg+0..7, 16+8*kg+0..7
__device__ __forceinline__ half16 ldA(const _Float16* row, int kg) {
  half16 r;
  ((half8*)&r)[0] = *(const half8*)(row + 8 * kg);
  ((half8*)&r)[1] = *(const half8*)(row + 16 + 8 * kg);
  return r;
}
// B-fragment 32x16 f16 from B^T row-major: lane holds col (lane&15); halves: k = 8*kg+0..7, 16+8*kg+0..7
__device__ __forceinline__ half16 ldB(const _Float16* row, int kg) {
  half16 r;
  ((half8*)&r)[0] = *(const half8*)(row + 8 * kg);
  ((half8*)&r)[1] = *(const half8*)(row + 16 + 8 * kg);
  return r;
}

// ---------------------------------------------------------------------------
__global__ void f32_to_f16_kernel(const float* __restrict__ src,
                                  _Float16* __restrict__ dst, int n) {
  int i = blockIdx.x * blockDim.x + threadIdx.x;
  if (i < n) dst[i] = (_Float16)src[i];
}

// ---------------------------------------------------------------------------
// QKV GEMM: out[8192,2304] = xh[8192,768] * wh[2304,768]^T
// Each wave computes a 32x32 tile (2x2 WMMA tiles). Epilogue scatters into
// Q [B,H,N,D] (scaled by D^-0.5 = 0.125), K [B,H,N,D], V^T [B,H,D,N] as f16.
__global__ __launch_bounds__(256) void qkv_gemm_kernel(
    const _Float16* __restrict__ xh, const _Float16* __restrict__ wh,
    _Float16* __restrict__ Qh, _Float16* __restrict__ Kh,
    _Float16* __restrict__ Vth) {
  const int lane = threadIdx.x & 31;
  const int wave = blockIdx.x * 8 + (threadIdx.x >> 5);
  const int tn = wave % 72;             // 2304/32
  const int tm = wave / 72;             // 8192/32
  const int mbase = tm * 32, nbase = tn * 32;
  const int kg = lane >> 4, lm = lane & 15;

  const _Float16* a0p = xh + (size_t)(mbase + lm) * 768;
  const _Float16* a1p = a0p + 16 * 768;
  const _Float16* b0p = wh + (size_t)(nbase + lm) * 768;
  const _Float16* b1p = b0p + 16 * 768;

  float8 acc[2][2] = {};
  for (int k = 0; k < 768; k += 32) {
    half16 a0 = ldA(a0p + k, kg);
    half16 a1 = ldA(a1p + k, kg);
    half16 b0 = ldB(b0p + k, kg);
    half16 b1 = ldB(b1p + k, kg);
    acc[0][0] = wmma16(a0, b0, acc[0][0]);
    acc[0][1] = wmma16(a0, b1, acc[0][1]);
    acc[1][0] = wmma16(a1, b0, acc[1][0]);
    acc[1][1] = wmma16(a1, b1, acc[1][1]);
  }

  const int mloc = kg << 3;  // D-layout: lanes 16..31 hold rows 8..15
  for (int mt = 0; mt < 2; ++mt)
    for (int nt = 0; nt < 2; ++nt)
      for (int i = 0; i < 8; ++i) {
        int m = mbase + 16 * mt + mloc + i;
        int n = nbase + 16 * nt + lm;
        float v = acc[mt][nt][i];
        int b = m >> 10, nseq = m & 1023;
        int t = n / 768, rem = n - t * 768;
        int h = rem >> 6, d = rem & 63;
        int bh = b * 12 + h;
        if (t == 0)
          Qh[((size_t)bh * 1024 + nseq) * 64 + d] = (_Float16)(v * 0.125f);
        else if (t == 1)
          Kh[((size_t)bh * 1024 + nseq) * 64 + d] = (_Float16)v;
        else
          Vth[((size_t)bh * 64 + d) * 1024 + nseq] = (_Float16)v;
      }
}

// ---------------------------------------------------------------------------
// Flash attention: block = (16 query-blocks, 96 bh-pairs), 4 waves/block,
// each wave owns 16 query rows, streams 32 keys/iteration.
__global__ __launch_bounds__(128) void flash_attn_kernel(
    const _Float16* __restrict__ Qh, const _Float16* __restrict__ Kh,
    const _Float16* __restrict__ Vth, _Float16* __restrict__ attnh) {
  __shared__ _Float16 plds[4][16 * 32];  // per-wave P-tile re-striping buffer

  const int lane = threadIdx.x & 31;
  const int wv = threadIdx.x >> 5;
  const int bh = blockIdx.y;
  const int q0 = blockIdx.x * 64 + wv * 16;
  const int kg = lane >> 4, lm = lane & 15;

  const _Float16* qrow = Qh + ((size_t)bh * 1024 + q0 + lm) * 64;
  half16 qa0 = ldA(qrow, kg);        // d 0..31 (pre-scaled by 0.125)
  half16 qa1 = ldA(qrow + 32, kg);   // d 32..63

  float8 O[4] = {};
  float mr[8], lr[8], cr[8], p0[8], p1[8];
  for (int i = 0; i < 8; ++i) { mr[i] = -3.0e38f; lr[i] = 0.f; }

  const _Float16* kbasep = Kh + (size_t)bh * 1024 * 64;
  const _Float16* vbasep = Vth + (size_t)bh * 64 * 1024;
  _Float16* pl = plds[wv];
  const int mb = kg << 3;

  for (int kb = 0; kb < 1024; kb += 32) {
    // --- S = Q * K^T for 32 keys (two 16x16 D-tiles) ---
    const _Float16* kr0 = kbasep + (size_t)(kb + lm) * 64;
    const _Float16* kr1 = kr0 + 16 * 64;
    half16 kb00 = ldB(kr0, kg);       // keys kb..kb+15, d 0..31
    half16 kb01 = ldB(kr0 + 32, kg);  // d 32..63
    half16 kb10 = ldB(kr1, kg);       // keys kb+16..kb+31
    half16 kb11 = ldB(kr1 + 32, kg);
    if (kb + 32 < 1024) __builtin_prefetch((const void*)(kr0 + 32 * 64), 0, 0);

    float8 s0 = {}, s1 = {};
    s0 = wmma16(qa0, kb00, s0);
    s0 = wmma16(qa1, kb01, s0);
    s1 = wmma16(qa0, kb10, s1);
    s1 = wmma16(qa1, kb11, s1);

    // --- online softmax update (row reductions across the 16-lane group) ---
    for (int i = 0; i < 8; ++i) {
      float v = fmaxf(s0[i], s1[i]);
      for (int off = 1; off < 16; off <<= 1) v = fmaxf(v, __shfl_xor(v, off, 32));
      float mn = fmaxf(mr[i], v);
      float c = __expf(mr[i] - mn);
      float e0 = __expf(s0[i] - mn);
      float e1 = __expf(s1[i] - mn);
      float rs = e0 + e1;
      for (int off = 1; off < 16; off <<= 1) rs += __shfl_xor(rs, off, 32);
      lr[i] = lr[i] * c + rs;
      mr[i] = mn;
      cr[i] = c;
      p0[i] = e0;
      p1[i] = e1;
    }
    for (int t = 0; t < 4; ++t)
      for (int i = 0; i < 8; ++i) O[t][i] *= cr[i];

    // --- re-stripe P (D-layout -> A-fragment) through LDS ---
    for (int i = 0; i < 8; ++i) {
      pl[(mb + i) * 32 + lm] = (_Float16)p0[i];
      pl[(mb + i) * 32 + 16 + lm] = (_Float16)p1[i];
    }
    __builtin_amdgcn_wave_barrier();
    half16 pf = ldA(pl + lm * 32, kg);
    __builtin_amdgcn_wave_barrier();

    // --- O += P * V  (V^T rows are contiguous along keys) ---
    for (int t = 0; t < 4; ++t) {
      const _Float16* vr = vbasep + (size_t)(16 * t + lm) * 1024 + kb;
      half16 vb = ldB(vr, kg);
      O[t] = wmma16(pf, vb, O[t]);
    }
  }

  // --- normalize and write attn output as f16 [B,N,C] ---
  const int b = bh / 12, h = bh - b * 12;
  for (int t = 0; t < 4; ++t)
    for (int i = 0; i < 8; ++i) {
      int m = q0 + mb + i;
      int d = 16 * t + lm;
      attnh[((size_t)b * 1024 + m) * 768 + h * 64 + d] =
          (_Float16)(O[t][i] / lr[i]);
    }
}

// ---------------------------------------------------------------------------
// Output projection: out[8192,768] = attn[8192,768] * proj_w[768,768]^T + b
__global__ __launch_bounds__(256) void proj_gemm_kernel(
    const _Float16* __restrict__ ah, const _Float16* __restrict__ wh,
    const float* __restrict__ bias, float* __restrict__ out) {
  const int lane = threadIdx.x & 31;
  const int wave = blockIdx.x * 8 + (threadIdx.x >> 5);
  const int tn = wave % 24;  // 768/32
  const int tm = wave / 24;  // 8192/32
  const int mbase = tm * 32, nbase = tn * 32;
  const int kg = lane >> 4, lm = lane & 15;

  const _Float16* a0p = ah + (size_t)(mbase + lm) * 768;
  const _Float16* a1p = a0p + 16 * 768;
  const _Float16* b0p = wh + (size_t)(nbase + lm) * 768;
  const _Float16* b1p = b0p + 16 * 768;

  float8 acc[2][2] = {};
  for (int k = 0; k < 768; k += 32) {
    half16 a0 = ldA(a0p + k, kg);
    half16 a1 = ldA(a1p + k, kg);
    half16 b0 = ldB(b0p + k, kg);
    half16 b1 = ldB(b1p + k, kg);
    acc[0][0] = wmma16(a0, b0, acc[0][0]);
    acc[0][1] = wmma16(a0, b1, acc[0][1]);
    acc[1][0] = wmma16(a1, b0, acc[1][0]);
    acc[1][1] = wmma16(a1, b1, acc[1][1]);
  }

  const int mloc = kg << 3;
  for (int mt = 0; mt < 2; ++mt)
    for (int nt = 0; nt < 2; ++nt)
      for (int i = 0; i < 8; ++i) {
        int m = mbase + 16 * mt + mloc + i;
        int n = nbase + 16 * nt + lm;
        out[(size_t)m * 768 + n] = acc[mt][nt][i] + bias[n];
      }
}

// ---------------------------------------------------------------------------
extern "C" void kernel_launch(void* const* d_in, const int* in_sizes, int n_in,
                              void* d_out, int out_size, void* d_ws,
                              size_t ws_size, hipStream_t stream) {
  (void)in_sizes; (void)n_in; (void)out_size; (void)ws_size;

  const float* x = (const float*)d_in[0];       // [8,1024,768]
  const float* qkv_w = (const float*)d_in[1];   // [2304,768]
  const float* proj_w = (const float*)d_in[2];  // [768,768]
  const float* proj_b = (const float*)d_in[3];  // [768]
  float* out = (float*)d_out;                   // [8,1024,768]

  const size_t NX = 8192ull * 768;   // x / attn-out elements
  const size_t NWQ = 2304ull * 768;  // qkv_w elements
  const size_t NWP = 768ull * 768;   // proj_w elements
  const size_t NHD = 8ull * 12 * 1024 * 64;  // per Q/K/V buffer

  _Float16* ws = (_Float16*)d_ws;    // ~67.6 MB of f16 total
  _Float16* xh = ws;
  _Float16* wqh = xh + NX;
  _Float16* wph = wqh + NWQ;
  _Float16* Qh = wph + NWP;
  _Float16* Kh = Qh + NHD;
  _Float16* Vth = Kh + NHD;
  _Float16* atth = Vth + NHD;

  f32_to_f16_kernel<<<(int)(NX / 256), 256, 0, stream>>>(x, xh, (int)NX);
  f32_to_f16_kernel<<<(int)(NWQ / 256), 256, 0, stream>>>(qkv_w, wqh, (int)NWQ);
  f32_to_f16_kernel<<<(int)(NWP / 256), 256, 0, stream>>>(proj_w, wph, (int)NWP);

  // 18432 wave-tiles of 32x32 over [8192,2304] -> 2304 blocks x 8 waves
  qkv_gemm_kernel<<<2304, 256, 0, stream>>>(xh, wqh, Qh, Kh, Vth);

  // 96 (b,h) pairs x 16 query-blocks of 64 rows, 4 waves/block
  flash_attn_kernel<<<dim3(16, 96), 128, 0, stream>>>(Qh, Kh, Vth, atth);

  // 6144 wave-tiles over [8192,768] -> 768 blocks x 8 waves
  proj_gemm_kernel<<<768, 256, 0, stream>>>(atth, wph, proj_b, out);
}
